// SelfExtendAttn_3564822855719
// MI455X (gfx1250) — matrix operand order repaired and checked by reading, not compiled, not run on hardware
//
#include <hip/hip_runtime.h>
#include <math.h>

// ---------------------------------------------------------------------------
// Self-extend attention for MI455X (gfx1250, wave32, WMMA bf16 16x16x32).
// B=2, S=4096, D=1024, G_SIZE=16, W_SIZE=512 (shift = 512 - 512/16 = 480).
//
// Pipeline:
//   1) pe_prep:    NQ = q+pe[i], GQ = q+pe[i/16+480], NK = k+pe[i], GK = k+pe[i/16]
//                  (bf16, [B,S,D]) in workspace.
//   2) v_transpose: VT[B,D,S] bf16 so PV B-operands are contiguous along keys.
//   3) attn:       flash attention, 32 query rows / block, 64-key tiles,
//                  per-tile variant selection, online softmax, WMMA QK + PV.
// Workspace requirement: 5 * B*S*D * 2 bytes = 80 MB.
// ---------------------------------------------------------------------------

#define B_LEN 2
#define S_LEN 4096
#define D_LEN 1024
#define W_SIZE 512

typedef __bf16 bf16;
typedef __attribute__((ext_vector_type(2)))  bf16  v2bf;
typedef __attribute__((ext_vector_type(8)))  bf16  v8bf;
typedef __attribute__((ext_vector_type(16))) bf16  v16bf;
typedef __attribute__((ext_vector_type(8)))  float v8f;

__device__ __forceinline__ v8f wmma_bf16(v16bf a, v16bf b, v8f c) {
  // D = A(16x32 bf16) x B(32x16 bf16) + C(16x16 f32)
  return __builtin_amdgcn_wmma_f32_16x16x32_bf16(
      /*neg_a=*/false, a, /*neg_b=*/false, b,
      /*c_mod=*/(short)0, c, /*reuse_a=*/false, /*reuse_b=*/false);
}

// A-matrix 16x32 bf16 fragment: lane half h, element e -> K=16h+e.
// Row data is contiguous, so two 16B loads.
__device__ __forceinline__ v16bf load_a16x32(const bf16* row_base, int h) {
  v8bf lo = *(const v8bf*)(row_base + 16 * h);
  v8bf hi = *(const v8bf*)(row_base + 16 * h + 8);
  v16bf r;
#pragma unroll
  for (int e = 0; e < 8; ++e) { r[e] = lo[e]; r[e + 8] = hi[e]; }
  return r;
}

// ---------------------------------------------------------------------------
// Kernel 1: fold sinusoidal PE into bf16 Q/K variants.
// grid = B*S blocks, 256 threads; each thread handles 2 (sin,cos) dim pairs.
// ---------------------------------------------------------------------------
__global__ void pe_prep(const float* __restrict__ q, const float* __restrict__ k,
                        const int* __restrict__ pos,
                        bf16* __restrict__ NQ, bf16* __restrict__ GQ,
                        bf16* __restrict__ NK, bf16* __restrict__ GK) {
  const int bs = blockIdx.x;           // 0 .. B*S-1
  const int s  = bs & (S_LEN - 1);
  const int p  = pos[s];               // reference position input
  const int gi = p >> 4;               // pos // 16
  const float fi = (float)p;
  const float fg = (float)gi;
  const float fs = (float)(gi + (W_SIZE - W_SIZE / 16));  // +480
  const size_t row = (size_t)bs * D_LEN;
  const float2* qr = (const float2*)(q + row);
  const float2* kr = (const float2*)(k + row);
  // div_term[t] = exp(-(2t) * ln(10000) / D); ln(10000)/512 = 0.01798894700...
#pragma unroll
  for (int u = 0; u < 2; ++u) {
    const int pr = threadIdx.x * 2 + u;          // pair index 0..511
    const float div = __expf(-0.0179889470f * (float)pr);
    float sn_n, cs_n, sn_g, cs_g, sn_s, cs_s;
    __sincosf(fi * div, &sn_n, &cs_n);
    __sincosf(fg * div, &sn_g, &cs_g);
    __sincosf(fs * div, &sn_s, &cs_s);
    const float2 qv = qr[pr];
    const float2 kv = kr[pr];
    v2bf o;
    o.x = (bf16)(qv.x + sn_n); o.y = (bf16)(qv.y + cs_n);
    ((v2bf*)(NQ + row))[pr] = o;
    o.x = (bf16)(qv.x + sn_s); o.y = (bf16)(qv.y + cs_s);
    ((v2bf*)(GQ + row))[pr] = o;
    o.x = (bf16)(kv.x + sn_n); o.y = (bf16)(kv.y + cs_n);
    ((v2bf*)(NK + row))[pr] = o;
    o.x = (bf16)(kv.x + sn_g); o.y = (bf16)(kv.y + cs_g);
    ((v2bf*)(GK + row))[pr] = o;
  }
}

// ---------------------------------------------------------------------------
// Kernel 2: VT[b][d][s] = (bf16)v[b][s][d], 64x64 LDS-tiled transpose.
// ---------------------------------------------------------------------------
__global__ void v_transpose(const float* __restrict__ v, bf16* __restrict__ VT) {
  __shared__ bf16 tile[64][64];
  const int b  = blockIdx.z;
  const int i0 = blockIdx.x * 64;
  const int d0 = blockIdx.y * 64;
  const int t  = threadIdx.x;
  {
    const int r  = t >> 2;
    const int c0 = (t & 3) * 16;
    const float4* src =
        (const float4*)(v + ((size_t)(b * S_LEN + i0 + r)) * D_LEN + d0 + c0);
#pragma unroll
    for (int e4 = 0; e4 < 4; ++e4) {
      float4 f = src[e4];
      tile[c0 + e4 * 4 + 0][r] = (bf16)f.x;
      tile[c0 + e4 * 4 + 1][r] = (bf16)f.y;
      tile[c0 + e4 * 4 + 2][r] = (bf16)f.z;
      tile[c0 + e4 * 4 + 3][r] = (bf16)f.w;
    }
  }
  __syncthreads();
  {
    const int dl = t >> 2;
    const int j0 = (t & 3) * 16;
    v16bf o;
#pragma unroll
    for (int e = 0; e < 16; ++e) o[e] = tile[dl][j0 + e];
    *(v16bf*)(VT + ((size_t)(b * D_LEN + d0 + dl)) * S_LEN + i0 + j0) = o;
  }
}

// ---------------------------------------------------------------------------
// Kernel 3: flash attention with variant-select score merge.
// grid = (S/32, B), 256 threads = 8 wave32.
//   QK: wave (rt = w>>2, ct = w&3) computes one 16x16 score subtile of the
//       32x64 score block, reducing all of D with 32 bf16 WMMA k-steps.
//   PV: wave w owns output D-chunk [128w, 128w+128): 2x8 f32 C fragments,
//       B fragments double-buffered so loads lead the WMMAs by one step.
// ---------------------------------------------------------------------------
__global__ __launch_bounds__(256, 1)
void attn(const bf16* __restrict__ NQ, const bf16* __restrict__ GQ,
          const bf16* __restrict__ NK, const bf16* __restrict__ GK,
          const bf16* __restrict__ VT, float* __restrict__ out) {
  __shared__ float s_sc[32][65];   // padded: conflict-free row walks
  __shared__ float s_m[32], s_l[32], s_scale[32];

  const int i0   = blockIdx.x * 32;
  const int b    = blockIdx.y;
  const int tid  = threadIdx.x;
  const int wave = tid >> 5;
  const int lane = tid & 31;
  const int ln16 = lane & 15;
  const int h    = lane >> 4;
  const int rtq  = wave >> 2;      // QK subtile row (0/1)
  const int ctq  = wave & 3;       // QK subtile col (0..3)
  const int d0w  = wave * 128;     // PV output column chunk

  if (tid < 32) { s_m[tid] = -__builtin_inff(); s_l[tid] = 0.f; }

  v8f acc[2][8] = {};              // O accumulators: 32 rows x 128 cols / wave

  // Per-lane A row base for QK (query row for this lane inside the subtile).
  const size_t qrow  = (size_t)(b * S_LEN + i0 + rtq * 16 + ln16) * D_LEN;
  const size_t krow0 = (size_t)(b * S_LEN + ctq * 16 + ln16) * D_LEN;

  const int ntiles = (i0 + 31) / 64 + 1;
  for (int t = 0; t < ntiles; ++t) {
    const int j0 = t * 64;
    // Fully-grouped: every causal (i,j) in tile has j <= i-512.
    const bool fullG = (j0 + 63) <= (i0 - W_SIZE);
    // Fully-neighbor: every causal (i,j) in tile has j >= i-511.
    const bool fullN = j0 >= (i0 + 31 - (W_SIZE - 1));
    const bool doN = !fullG, doG = !fullN;

    __syncthreads();               // previous tile's s_sc readers are done

    // -------- QK: this wave's 16x16 subtile over full D --------------------
    {
      const size_t krow = krow0 + (size_t)j0 * D_LEN;
      if (j0 + 64 <= i0 + 31) {    // prefetch next key tile into L2
        __builtin_prefetch(NK + krow + (size_t)64 * D_LEN, 0, 1);
        __builtin_prefetch(GK + krow + (size_t)64 * D_LEN, 0, 1);
      }
      v8f cN = {}, cG = {};
#pragma unroll 4
      for (int ks = 0; ks < 32; ++ks) {
        const int dk = ks * 32;
        if (doN) {
          v16bf a  = load_a16x32(NQ + qrow + dk, h);
          v16bf bm = *(const v16bf*)(NK + krow + dk + 16 * h);
          cN = wmma_bf16(a, bm, cN);
        }
        if (doG) {
          v16bf a  = load_a16x32(GQ + qrow + dk, h);
          v16bf bm = *(const v16bf*)(GK + krow + dk + 16 * h);
          cG = wmma_bf16(a, bm, cG);
        }
      }
      // Per-element variant select, plain store (8 disjoint subtiles).
#pragma unroll
      for (int v = 0; v < 8; ++v) {
        const int row = rtq * 16 + v + 8 * h;    // C layout: M = v + 8*(lane/16)
        const int col = ctq * 16 + ln16;
        const int i = i0 + row, j = j0 + col;
        float val;
        if (doN && doG) val = (j >= i - (W_SIZE - 1)) ? cN[v] : cG[v];
        else            val = doN ? cN[v] : cG[v];
        s_sc[row][col] = val;
      }
    }
    __syncthreads();

    // -------- online softmax stats (wave 0, one row per lane) --------------
    // Writes masked exp(s - m_new) back into s_sc, so the P-fragment build
    // below is a pure bf16 convert (no per-lane exp/mask duplication).
    if (tid < 32) {
      const int r = tid, i = i0 + r;
      float mx = -__builtin_inff();
      for (int c = 0; c < 64; ++c)
        if (j0 + c <= i) mx = fmaxf(mx, s_sc[r][c]);
      const float mold = s_m[r];
      const float mnew = fmaxf(mold, mx);
      const float scale = (mnew == -__builtin_inff()) ? 1.f : __expf(mold - mnew);
      float sum = 0.f;
      for (int c = 0; c < 64; ++c) {
        const float p = (j0 + c <= i) ? __expf(s_sc[r][c] - mnew) : 0.f;
        s_sc[r][c] = p;
        sum += p;
      }
      s_l[r] = s_l[r] * scale + sum;
      s_m[r] = mnew;
      s_scale[r] = scale;
    }
    __syncthreads();

    // -------- rescale O accumulators ---------------------------------------
#pragma unroll
    for (int rt = 0; rt < 2; ++rt)
#pragma unroll
      for (int v = 0; v < 8; ++v) {
        const float sc = s_scale[rt * 16 + v + 8 * h];
#pragma unroll
        for (int ct = 0; ct < 8; ++ct) acc[rt][ct][v] *= sc;
      }

    // -------- P fragments (bf16 A-layout 16x32): pure converts -------------
    v16bf P[2][2];
#pragma unroll
    for (int rt = 0; rt < 2; ++rt) {
      const int row = rt * 16 + ln16;
#pragma unroll
      for (int kk = 0; kk < 2; ++kk) {
        v16bf pf;
#pragma unroll
        for (int e = 0; e < 16; ++e) {
          const int kl = 16 * h + e;
          pf[e] = (bf16)s_sc[row][kk * 32 + kl];
        }
        P[rt][kk] = pf;
      }
    }

    // -------- PV: O += P (32x64) x V (64x128 chunk), double-buffered -------
    {
      const bf16* vbase =
          VT + (size_t)(b * D_LEN + d0w + ln16) * S_LEN + j0 + 16 * h;
#define VFRAG(ct, kk) \
  (*(const v16bf*)(vbase + (size_t)(ct) * 16 * S_LEN + (kk) * 32))
      v16bf b0 = VFRAG(0, 0);
      v16bf b1 = VFRAG(0, 1);
#pragma unroll
      for (int ct = 0; ct < 8; ++ct) {
        v16bf n0 = b0, n1 = b1;
        if (ct < 7) { n0 = VFRAG(ct + 1, 0); n1 = VFRAG(ct + 1, 1); }
        acc[0][ct] = wmma_bf16(P[0][0], b0, acc[0][ct]);
        acc[1][ct] = wmma_bf16(P[1][0], b0, acc[1][ct]);
        acc[0][ct] = wmma_bf16(P[0][1], b1, acc[0][ct]);
        acc[1][ct] = wmma_bf16(P[1][1], b1, acc[1][ct]);
        b0 = n0; b1 = n1;
      }
#undef VFRAG
    }
  }

  __syncthreads();
  // -------- epilogue: normalize and store ----------------------------------
#pragma unroll
  for (int rt = 0; rt < 2; ++rt)
#pragma unroll
    for (int v = 0; v < 8; ++v) {
      const int row = rt * 16 + v + 8 * h;
      const float inv = 1.f / s_l[row];
      float* orow = out + (size_t)(b * S_LEN + i0 + row) * D_LEN + d0w + ln16;
#pragma unroll
      for (int ct = 0; ct < 8; ++ct) orow[ct * 16] = acc[rt][ct][v] * inv;
    }
}

// ---------------------------------------------------------------------------
extern "C" void kernel_launch(void* const* d_in, const int* in_sizes, int n_in,
                              void* d_out, int out_size, void* d_ws, size_t ws_size,
                              hipStream_t stream) {
  const float* q  = (const float*)d_in[0];
  const float* k  = (const float*)d_in[1];
  const float* v  = (const float*)d_in[2];
  const int* pos  = (const int*)d_in[3];
  float* out      = (float*)d_out;

  // Workspace layout (bf16): NQ | GQ | NK | GK | VT, each B*S*D elements.
  const size_t SZ = (size_t)B_LEN * S_LEN * D_LEN;     // 8,388,608 elements
  bf16* ws = (bf16*)d_ws;
  bf16* NQ = ws;
  bf16* GQ = ws + 1 * SZ;
  bf16* NK = ws + 2 * SZ;
  bf16* GK = ws + 3 * SZ;
  bf16* VT = ws + 4 * SZ;                              // needs 80 MB total

  pe_prep<<<dim3(B_LEN * S_LEN), 256, 0, stream>>>(q, k, pos, NQ, GQ, NK, GK);
  v_transpose<<<dim3(S_LEN / 64, D_LEN / 64, B_LEN), 256, 0, stream>>>(v, VT);
  attn<<<dim3(S_LEN / 32, B_LEN), 256, 0, stream>>>(NQ, GQ, NK, GK, VT, out);
}
